// GapModel_3728031613301
// MI455X (gfx1250) — matrix operand order repaired and checked by reading, not compiled, not run on hardware
//
#include <hip/hip_runtime.h>
#include <hip/hip_bf16.h>

typedef __attribute__((ext_vector_type(16))) __bf16 v16bf;
typedef __attribute__((ext_vector_type(8)))  __bf16 v8bf;
typedef __attribute__((ext_vector_type(8)))  float  v8f;

#define N_ENV    100000
#define D_PS     512
#define N_SPEC   4
#define N_SUP    256
#define N_STRUCT 1000

// LDS B-tile: 16 columns x 512 K bf16, row stride padded to 1040 B
// (260 dwords; 260 % 64 = 4 -> each half-wave's 16x b128 reads hit all
// 64 banks exactly once: conflict-free).
#define BROW   1040
#define BTILE  (16 * BROW)

// ---------------- workspace layout (bytes) ----------------
#define OFF_PSB   0
#define OFF_SPB   (OFF_PSB + (size_t)N_ENV * D_PS * 2)
#define OFF_PERR  (OFF_SPB + (size_t)N_SPEC * N_SUP * D_PS * 2)
#define OFF_IDX   (OFF_PERR + (size_t)N_ENV * 4)
#define OFF_META  (OFF_IDX + (size_t)N_ENV * 4)

#define WAIT_ASYNC() asm volatile("s_wait_asynccnt 0x0" ::: "memory")

__global__ void init_meta(int* meta) {
    if (threadIdx.x < 12) meta[threadIdx.x] = 0;
}

// Convert support points f32 -> bf16 (already unit norm). 1024 blocks x 256 thr.
__global__ void prep_support(const float* __restrict__ sp, __bf16* __restrict__ spb) {
    size_t i = (size_t)blockIdx.x * D_PS + (size_t)threadIdx.x * 2;
    spb[i]     = (__bf16)sp[i];
    spb[i + 1] = (__bf16)sp[i + 1];
}

// Normalize one env row, convert to bf16, count species. 100000 blocks x 256 thr.
__global__ void prep_env(const float* __restrict__ ps, const int* __restrict__ species,
                         __bf16* __restrict__ psb, int* __restrict__ counts) {
    const int row = blockIdx.x;
    const float2 x = ((const float2*)(ps + (size_t)row * D_PS))[threadIdx.x];
    __shared__ float red[256];
    red[threadIdx.x] = x.x * x.x + x.y * x.y;
    __syncthreads();
    for (int o = 128; o > 0; o >>= 1) {
        if (threadIdx.x < o) red[threadIdx.x] += red[threadIdx.x + o];
        __syncthreads();
    }
    const float inv = rsqrtf(red[0]);
    __bf16* dst = psb + (size_t)row * D_PS;
    dst[2 * threadIdx.x]     = (__bf16)(x.x * inv);
    dst[2 * threadIdx.x + 1] = (__bf16)(x.y * inv);
    if (threadIdx.x == 0) atomicAdd(&counts[species[row]], 1);
}

// meta: [0..3]=counts, [4..7]=offs, [8..11]=cursors
__global__ void scan_meta(int* meta) {
    int acc = 0;
    for (int s = 0; s < N_SPEC; ++s) {
        meta[4 + s] = acc;
        meta[8 + s] = acc;
        acc += meta[s];
    }
}

// Scatter row indices into species buckets (block-aggregated atomics).
__global__ void scatter_rows(const int* __restrict__ species, int* __restrict__ meta,
                             int* __restrict__ idxbuf) {
    const int row = blockIdx.x * blockDim.x + threadIdx.x;
    __shared__ int lcnt[N_SPEC];
    __shared__ int lbase[N_SPEC];
    if (threadIdx.x < N_SPEC) lcnt[threadIdx.x] = 0;
    __syncthreads();
    int s = 0, mypos = 0;
    if (row < N_ENV) {
        s = species[row];
        mypos = atomicAdd(&lcnt[s], 1);
    }
    __syncthreads();
    if (threadIdx.x < N_SPEC)
        lbase[threadIdx.x] = atomicAdd(&meta[8 + threadIdx.x], lcnt[threadIdx.x]);
    __syncthreads();
    if (row < N_ENV) idxbuf[lbase[s] + mypos] = row;
}

// Async-stage one 16-col x 512-K bf16 B tile (16 KB) into LDS.
// 256 threads x 4 chunks x 16 B, fully coalesced, tracked by ASYNCcnt.
__device__ __forceinline__ void stage_tile(const char* gbase, char* lbuf, int tid) {
#pragma unroll
    for (int j = 0; j < 4; ++j) {
        const int c    = tid + 256 * j;        // 0..1023 chunk id
        const int row  = c >> 6;               // 0..15 column
        const int byte = (c & 63) << 4;        // 0..1008 within column
        const unsigned lds = (unsigned)(size_t)(lbuf + row * BROW + byte);
        const unsigned long long ga =
            (unsigned long long)(size_t)(gbase + (size_t)row * (D_PS * 2) + byte);
        asm volatile("global_load_async_to_lds_b128 %0, %1, off"
                     :: "v"(lds), "v"(ga) : "memory");
    }
}

__device__ __forceinline__ v16bf loadB(const char* bcol, int kt) {
    v8bf b0 = *(const v8bf*)(bcol + kt * 64);
    v8bf b1 = *(const v8bf*)(bcol + kt * 64 + 32);
    v16bf b;
#pragma unroll
    for (int i = 0; i < 8; ++i) { b[i] = b0[i]; b[i + 8] = b1[i]; }
    return b;
}

// WMMA GEMM: grid = (391, 4), block = 256 (8 waves). Each wave: 32 gathered
// rows x 256 support cols x K=512. Two register-resident A panels share each
// LDS B fragment (2 WMMAs per 1 KB of LDS reads -> 512 B/WMMA, matching the
// 256 B/clk WGP LDS bandwidth at 4 concurrent SIMDs). B fragments run in a
// 4-deep register pipeline so the k-loop never drains DScnt; B tiles are
// double-buffered in LDS via global_load_async_to_lds_b128.
__global__ void __launch_bounds__(256)
gemm_energy(const __bf16* __restrict__ psb, const __bf16* __restrict__ spb,
            const float* __restrict__ w, const int* __restrict__ idxbuf,
            const int* __restrict__ meta, float* __restrict__ per_row) {
    __shared__ __align__(16) char btile[2][BTILE];

    const int s    = blockIdx.y;
    const int cnt  = meta[s];
    const int off  = meta[4 + s];
    if (blockIdx.x * 256 >= cnt) return;       // uniform block-level exit only

    const int tid  = threadIdx.x;
    const int wave = tid >> 5;
    const int lane = tid & 31;
    const int base = blockIdx.x * 256 + wave * 32;
    const int half = lane >> 4;
    const int lrow = lane & 15;

    // Gather rows for this lane's two A row-groups (clamped; stores predicated).
    int p0 = base + lrow;       if (p0 > cnt - 1) p0 = cnt - 1;
    int p1 = base + 16 + lrow;  if (p1 > cnt - 1) p1 = cnt - 1;
    const __bf16* arow0 = psb + (size_t)idxbuf[off + p0] * D_PS;
    const __bf16* arow1 = psb + (size_t)idxbuf[off + p1] * D_PS;

    // Two register-resident 16x512 A panels, ISA 16-bit A-layout per lane:
    // K = kb + half*8 + [0..7] and K = kb + 16 + half*8 + [0..7].
    v16bf A0[16], A1[16];
#pragma unroll
    for (int kt = 0; kt < 16; ++kt) {
        const int kb = kt * 32 + half * 8;
        v8bf l0 = *(const v8bf*)(arow0 + kb);
        v8bf h0 = *(const v8bf*)(arow0 + kb + 16);
        v8bf l1 = *(const v8bf*)(arow1 + kb);
        v8bf h1 = *(const v8bf*)(arow1 + kb + 16);
#pragma unroll
        for (int i = 0; i < 8; ++i) {
            A0[kt][i] = l0[i]; A0[kt][i + 8] = h0[i];
            A1[kt][i] = l1[i]; A1[kt][i + 8] = h1[i];
        }
    }

    float rs0[8], rs1[8];
#pragma unroll
    for (int v = 0; v < 8; ++v) { rs0[v] = 0.f; rs1[v] = 0.f; }

    const char* gsp = (const char*)(spb + (size_t)s * N_SUP * D_PS);

    // Prologue: stage tile 0.
    stage_tile(gsp, &btile[0][0], tid);
    WAIT_ASYNC();
    __syncthreads();

    for (int nt = 0; nt < 16; ++nt) {
        const char* cur = &btile[nt & 1][0];
        if (nt + 1 < 16)
            stage_tile(gsp + (size_t)(nt + 1) * 16 * (D_PS * 2),
                       &btile[(nt + 1) & 1][0], tid);

        // B fragments from LDS: column nt*16 + lrow, K = half*8 + [0..7] and
        // 16 + half*8 + [0..7] per lane, 4-deep register pipeline.
        const char* bcol = cur + lrow * BROW + half * 16;
        v16bf B[4];
#pragma unroll
        for (int i = 0; i < 4; ++i) B[i] = loadB(bcol, i);

        v8f c0 = {}, c1 = {};
#pragma unroll
        for (int kt = 0; kt < 16; ++kt) {
            c0 = __builtin_amdgcn_wmma_f32_16x16x32_bf16(
                     false, A0[kt], false, B[kt & 3], (short)0, c0, false, false);
            c1 = __builtin_amdgcn_wmma_f32_16x16x32_bf16(
                     false, A1[kt], false, B[kt & 3], (short)0, c1, false, false);
            if (kt + 4 < 16) B[kt & 3] = loadB(bcol, kt + 4);
        }
        const float wcol = w[(size_t)s * N_SUP + nt * 16 + lrow];
#pragma unroll
        for (int v = 0; v < 8; ++v) {
            rs0[v] += wcol * c0[v] * c0[v];
            rs1[v] += wcol * c1[v] * c1[v];
        }

        if (nt + 1 < 16) {
            WAIT_ASYNC();        // staged data for nt+1 resident in LDS
            __syncthreads();     // all waves done reading tile nt's buffer peer
        }
    }

    // Reduce across the 16 lanes of each half-wave (columns) -> row totals.
#pragma unroll
    for (int v = 0; v < 8; ++v) {
        rs0[v] += __shfl_xor(rs0[v], 1, 16);
        rs0[v] += __shfl_xor(rs0[v], 2, 16);
        rs0[v] += __shfl_xor(rs0[v], 4, 16);
        rs0[v] += __shfl_xor(rs0[v], 8, 16);
        rs1[v] += __shfl_xor(rs1[v], 1, 16);
        rs1[v] += __shfl_xor(rs1[v], 2, 16);
        rs1[v] += __shfl_xor(rs1[v], 4, 16);
        rs1[v] += __shfl_xor(rs1[v], 8, 16);
    }

    if (lrow == 0) {
#pragma unroll
        for (int v = 0; v < 8; ++v) {
            const int q0 = base + half * 8 + v;        // rows 0..15 group
            const int q1 = base + 16 + half * 8 + v;   // rows 16..31 group
            if (q0 < cnt) per_row[idxbuf[off + q0]] = rs0[v];
            if (q1 < cnt) per_row[idxbuf[off + q1]] = rs1[v];
        }
    }
}

// Deterministic segment sum: one block per structure, binary search sorted ids.
__global__ void segsum(const int* __restrict__ sid, const float* __restrict__ per_row,
                       float* __restrict__ out) {
    const int t = blockIdx.x;
    int lo = 0, hi = N_ENV;
    while (lo < hi) { int m = (lo + hi) >> 1; if (sid[m] < t) lo = m + 1; else hi = m; }
    const int s0 = lo;
    hi = N_ENV;
    while (lo < hi) { int m = (lo + hi) >> 1; if (sid[m] < t + 1) lo = m + 1; else hi = m; }
    const int s1 = lo;
    float acc = 0.f;
    for (int i = s0 + threadIdx.x; i < s1; i += blockDim.x) acc += per_row[i];
    __shared__ float red[128];
    red[threadIdx.x] = acc;
    __syncthreads();
    for (int o = 64; o > 0; o >>= 1) {
        if (threadIdx.x < o) red[threadIdx.x] += red[threadIdx.x + o];
        __syncthreads();
    }
    if (threadIdx.x == 0) out[t] = red[0];
}

extern "C" void kernel_launch(void* const* d_in, const int* in_sizes, int n_in,
                              void* d_out, int out_size, void* d_ws, size_t ws_size,
                              hipStream_t stream) {
    const float* ps      = (const float*)d_in[0];
    const float* sp      = (const float*)d_in[1];
    const float* weights = (const float*)d_in[2];
    const int*   species = (const int*)d_in[3];
    const int*   sid     = (const int*)d_in[4];
    float* out = (float*)d_out;

    char* ws = (char*)d_ws;
    __bf16* psb    = (__bf16*)(ws + OFF_PSB);
    __bf16* spb    = (__bf16*)(ws + OFF_SPB);
    float*  perr   = (float*)(ws + OFF_PERR);
    int*    idxbuf = (int*)(ws + OFF_IDX);
    int*    meta   = (int*)(ws + OFF_META);

    init_meta<<<1, 32, 0, stream>>>(meta);
    prep_support<<<N_SPEC * N_SUP, 256, 0, stream>>>(sp, spb);
    prep_env<<<N_ENV, 256, 0, stream>>>(ps, species, psb, meta);
    scan_meta<<<1, 1, 0, stream>>>(meta);
    scatter_rows<<<(N_ENV + 255) / 256, 256, 0, stream>>>(species, meta, idxbuf);
    gemm_energy<<<dim3((N_ENV + 255) / 256, N_SPEC), 256, 0, stream>>>(
        psb, spb, weights, idxbuf, meta, perr);
    segsum<<<N_STRUCT, 128, 0, stream>>>(sid, perr, out);
}
